// IGITrans_46273977647533
// MI455X (gfx1250) — matrix operand order repaired and checked by reading, not compiled, not run on hardware
//
#include <hip/hip_runtime.h>
#include <hip/hip_bf16.h>
#include <stdint.h>

// ---------------------------------------------------------------------------
// Types for CDNA5 WMMA (wave32): bf16 operands, f32 accumulate
// ---------------------------------------------------------------------------
typedef __attribute__((ext_vector_type(16))) __bf16 v16bf;
typedef __attribute__((ext_vector_type(8)))  __bf16 v8bf;
typedef __attribute__((ext_vector_type(8)))  float  v8f;

#define H_IMG 128
#define W_IMG 128
#define NBATCH 64

// spatial sizes
#define OH1 134
#define OW1 134
#define OH2 132
#define OW2 132
#define OH3 130
#define OW3 130

// CDNA5 async global->LDS copy (ASYNCcnt path), guarded so compilation can't
// regress if this toolchain doesn't declare the builtins.
#if defined(__gfx1250__) && \
    __has_builtin(__builtin_amdgcn_global_load_async_to_lds_b128) && \
    __has_builtin(__builtin_amdgcn_s_wait_asynccnt)
#define USE_ASYNC_LDS 1
#else
#define USE_ASYNC_LDS 0
#endif

// ---------------------------------------------------------------------------
// Prep: fold conv-bias+BN into per-channel scale/shift, reorder conv2/conv3
// weights into [oc][tap*C_in + ic] bf16 GEMM layout.
// scl buffer layout (floats): [0..63]=scale1 [64..127]=shift1
//   [128..159]=scale2 [160..191]=shift2 [192..223]=scale3 [224..255]=shift3
// ---------------------------------------------------------------------------
__global__ __launch_bounds__(256) void prep_kernel(
    const float* __restrict__ w2, const float* __restrict__ w3,
    const float* __restrict__ b1, const float* __restrict__ g1,
    const float* __restrict__ be1, const float* __restrict__ m1, const float* __restrict__ v1,
    const float* __restrict__ b2, const float* __restrict__ g2,
    const float* __restrict__ be2, const float* __restrict__ m2, const float* __restrict__ v2,
    const float* __restrict__ b3, const float* __restrict__ g3,
    const float* __restrict__ be3, const float* __restrict__ m3, const float* __restrict__ v3,
    float* __restrict__ scl, __bf16* __restrict__ w2m, __bf16* __restrict__ w3m)
{
    const int tid = blockIdx.x * blockDim.x + threadIdx.x;
    const int nth = gridDim.x * blockDim.x;
    for (int c = tid; c < 64; c += nth) {
        float inv = g1[c] * rsqrtf(v1[c] + 1e-5f);
        scl[c]      = inv;
        scl[64 + c] = b1[c] * inv + be1[c] - m1[c] * inv;
    }
    for (int c = tid; c < 32; c += nth) {
        float inv2 = g2[c] * rsqrtf(v2[c] + 1e-5f);
        scl[128 + c] = inv2;
        scl[160 + c] = b2[c] * inv2 + be2[c] - m2[c] * inv2;
        float inv3 = g3[c] * rsqrtf(v3[c] + 1e-5f);
        scl[192 + c] = inv3;
        scl[224 + c] = b3[c] * inv3 + be3[c] - m3[c] * inv3;
    }
    // w2: [32,64,3,3] -> w2m[oc*576 + tap*64 + ic]
    for (int i = tid; i < 32 * 576; i += nth) {
        int oc = i / 576, r = i % 576;
        int tap = r / 64, ic = r % 64;
        w2m[i] = (__bf16)w2[(oc * 64 + ic) * 9 + tap];
    }
    // w3: [32,32,3,3] -> w3m[oc*288 + tap*32 + ic]
    for (int i = tid; i < 32 * 288; i += nth) {
        int oc = i / 288, r = i % 288;
        int tap = r / 32, ic = r % 32;
        w3m[i] = (__bf16)w3[(oc * 32 + ic) * 9 + tap];
    }
}

// ---------------------------------------------------------------------------
// conv1: 1->64 ch, 3x3, pad 4, BN+ReLU, write bf16 NHWC [64,134,134,64].
// Degenerate GEMM (K=9) -> scalar VALU. 9 input taps per thread, all 64 oc.
// ---------------------------------------------------------------------------
__global__ __launch_bounds__(256) void conv1_kernel(
    const float* __restrict__ x, const float* __restrict__ w1,
    const float* __restrict__ scl, __bf16* __restrict__ h1)
{
    __shared__ float wl[64 * 9];
    __shared__ float sl[64], shl[64];
    for (int i = threadIdx.x; i < 576; i += 256) wl[i] = w1[i];
    for (int i = threadIdx.x; i < 64; i += 256) { sl[i] = scl[i]; shl[i] = scl[64 + i]; }
    __syncthreads();

    int gid = blockIdx.x * 256 + threadIdx.x;
    if (gid >= NBATCH * OH1 * OW1) return;
    int b = gid / (OH1 * OW1);
    int p = gid % (OH1 * OW1);
    int y = p / OW1, xo = p % OW1;

    float v[9];
#pragma unroll
    for (int ky = 0; ky < 3; ky++)
#pragma unroll
        for (int kx = 0; kx < 3; kx++) {
            int iy = y + ky - 4, ix = xo + kx - 4;
            v[ky * 3 + kx] = (iy >= 0 && iy < H_IMG && ix >= 0 && ix < W_IMG)
                                 ? x[(b * H_IMG + iy) * W_IMG + ix] : 0.0f;
        }

    __bf16* dst = h1 + (long)gid * 64;
#pragma unroll
    for (int g = 0; g < 8; g++) {
        v8bf o;
#pragma unroll
        for (int r = 0; r < 8; r++) {
            int oc = g * 8 + r;
            float acc = 0.0f;
#pragma unroll
            for (int t = 0; t < 9; t++) acc += wl[oc * 9 + t] * v[t];
            acc = fmaxf(acc * sl[oc] + shl[oc], 0.0f);
            o[r] = (__bf16)acc;
        }
        *(v8bf*)(dst + g * 8) = o;
    }
}

// ---------------------------------------------------------------------------
// WMMA implicit-GEMM conv: C_IN -> 32 ch, 3x3, no pad, BN+ReLU, bf16 NHWC out.
// Per wave: 32(oc) x 16(pixels) tile; K = 9*C_IN in chunks of 32
// (k = tap*C_IN + ic). A = weights (LDS-staged via async-to-LDS), B = im2col
// activations (one contiguous 32-byte global load per lane per chunk).
// ---------------------------------------------------------------------------
template <int C_IN, int OW, int OH>
__global__ __launch_bounds__(256) void conv_wmma_kernel(
    const __bf16* __restrict__ in, const __bf16* __restrict__ wmat,
    const float* __restrict__ scale, const float* __restrict__ shift,
    __bf16* __restrict__ out)
{
    constexpr int IW = OW + 2, IH = OH + 2;
    constexpr int KTOT = 9 * C_IN;
    constexpr int KCH = KTOT / 32;
    constexpr int NPIX = OW * OH;

    __shared__ __bf16 wl[32 * KTOT];

    // ---- Stage GEMM weights into LDS --------------------------------------
#if USE_ASYNC_LDS
    {
        // global->LDS DMA, 16B per lane per issue; tracked with ASYNCcnt.
        // Builtin expects (int4 __device__* src, int4 __shared__* dst, imm, imm).
        typedef __attribute__((__vector_size__(16))) int v4i_t;
        typedef __attribute__((address_space(1))) v4i_t gv4i;
        typedef __attribute__((address_space(3))) v4i_t lv4i;
        gv4i* gsrc = (gv4i*)(unsigned long long)(const void*)wmat;
        lv4i* ldst = (lv4i*)(unsigned)(unsigned long long)(void*)wl;
        constexpr int CHUNKS = (32 * KTOT * 2) / 16;   // 16-byte chunks
        for (int i = threadIdx.x; i < CHUNKS; i += 256)
            __builtin_amdgcn_global_load_async_to_lds_b128(gsrc + i, ldst + i, 0, 0);
        __builtin_amdgcn_s_wait_asynccnt(0);  // drain this wave's async copies
    }
#else
    {
        const uint4* s = (const uint4*)wmat;
        uint4* d = (uint4*)wl;
        for (int i = threadIdx.x; i < (32 * KTOT) / 8; i += 256) d[i] = s[i];
    }
#endif
    __syncthreads();

    const int lane = threadIdx.x & 31;
    const int wave = threadIdx.x >> 5;
    const int b = blockIdx.y;
    const int tileBase = (blockIdx.x * 8 + wave) * 16;
    if (tileBase >= NPIX) return;

    const int col = lane & 15;    // output pixel within tile / matrix column
    const int half = lane >> 4;   // lane group 0/1
    const int p = tileBase + col;
    const int pc = p < NPIX ? p : NPIX - 1;  // clamp so EXEC stays all-ones
    const int y = pc / OW, x = pc % OW;

    const __bf16* inp = in + ((long)(b * IH + y) * IW + x) * C_IN + half * 16;
    const int c0 = half * 16;
    const __bf16* a0base = &wl[col * KTOT];
    const __bf16* a1base = &wl[(16 + col) * KTOT];

    v8f acc0 = {}, acc1 = {};
#pragma unroll
    for (int t = 0; t < KCH; t++) {
        const int kbase = t * 32;
        const int tap = (t * 32) / C_IN;
        const int icc = (t * 32) % C_IN;
        const int ky = tap / 3, kx = tap % 3;

        // B fragment: lane's pixel, 16 consecutive input channels (NHWC)
        v16bf bfrag = *(const v16bf*)(inp + (ky * IW + kx) * C_IN + icc);

        // A fragments from LDS: K {c0..c0+7} and {c0+8..c0+15} of this chunk
        v8bf a0lo = *(const v8bf*)(a0base + kbase + c0);
        v8bf a0hi = *(const v8bf*)(a0base + kbase + c0 + 8);
        v8bf a1lo = *(const v8bf*)(a1base + kbase + c0);
        v8bf a1hi = *(const v8bf*)(a1base + kbase + c0 + 8);
        v16bf afrag0 = __builtin_shufflevector(a0lo, a0hi, 0,1,2,3,4,5,6,7,8,9,10,11,12,13,14,15);
        v16bf afrag1 = __builtin_shufflevector(a1lo, a1hi, 0,1,2,3,4,5,6,7,8,9,10,11,12,13,14,15);

        acc0 = __builtin_amdgcn_wmma_f32_16x16x32_bf16(false, afrag0, false, bfrag,
                                                       (short)0, acc0, false, false);
        acc1 = __builtin_amdgcn_wmma_f32_16x16x32_bf16(false, afrag1, false, bfrag,
                                                       (short)0, acc1, false, false);
    }

    if (p < NPIX) {
        __bf16* dst = out + ((long)(b * OH + y) * OW + x) * 32;
#pragma unroll
        for (int mt = 0; mt < 2; mt++) {
            const int ocb = mt * 16 + half * 8;  // D layout: 8 consecutive oc per lane
            v8f acc = mt ? acc1 : acc0;
            v8bf o;
#pragma unroll
            for (int r = 0; r < 8; r++) {
                float v = acc[r] * scale[ocb + r] + shift[ocb + r];
                o[r] = (__bf16)fmaxf(v, 0.0f);
            }
            *(v8bf*)(dst + ocb) = o;
        }
    }
}

// ---------------------------------------------------------------------------
// conv4: 32->1 ch, 3x3, no pad -> heat_raw fp32 [64,128,128] (written to d_out)
// ---------------------------------------------------------------------------
__global__ __launch_bounds__(256) void conv4_kernel(
    const __bf16* __restrict__ h3, const float* __restrict__ w4,
    const float* __restrict__ b4, float* __restrict__ heat_raw)
{
    __shared__ float wl[288];  // [tap][ic]
    for (int i = threadIdx.x; i < 288; i += 256) {
        int tap = i / 32, ic = i % 32;
        wl[i] = w4[ic * 9 + tap];
    }
    __syncthreads();

    int gid = blockIdx.x * 256 + threadIdx.x;
    if (gid >= NBATCH * H_IMG * W_IMG) return;
    int b = gid / (H_IMG * W_IMG);
    int p = gid % (H_IMG * W_IMG);
    int y = p / W_IMG, xo = p % W_IMG;

    float acc = b4[0];
#pragma unroll
    for (int ky = 0; ky < 3; ky++)
#pragma unroll
        for (int kx = 0; kx < 3; kx++) {
            const __bf16* s = h3 + (((long)(b * OH3 + y + ky) * OW3) + (xo + kx)) * 32;
            int tap = ky * 3 + kx;
#pragma unroll
            for (int ic = 0; ic < 32; ic++) acc += wl[tap * 32 + ic] * (float)s[ic];
        }
    heat_raw[gid] = acc;
}

// ---------------------------------------------------------------------------
// Per-8x8-block: max(sigmoid), sum, sum*xref, sum*yref.  1 block per image.
// ---------------------------------------------------------------------------
__global__ __launch_bounds__(256) void pool_kernel(
    const float* __restrict__ heat_raw, float* __restrict__ gridv,
    float* __restrict__ sA, float* __restrict__ sxA, float* __restrict__ syA)
{
    const int b = blockIdx.x, t = threadIdx.x;
    const int gy = t / 16, gx = t % 16;
    float mx = -1e30f, s = 0.f, sx = 0.f, sy = 0.f;
    for (int ya = 0; ya < 8; ya++)
        for (int xc = 0; xc < 8; xc++) {
            int yy = gy * 8 + ya, xx = gx * 8 + xc;
            float hr = heat_raw[(b * H_IMG + yy) * W_IMG + xx];
            float v = 1.0f / (1.0f + __expf(-hr));
            mx = fmaxf(mx, v);
            s += v;
            sx += v * (xx * (1.0f / 127.0f));
            sy += v * (yy * (1.0f / 127.0f));
        }
    int idx = b * 256 + t;
    gridv[idx] = mx; sA[idx] = s; sxA[idx] = sx; syA[idx] = sy;
}

// ---------------------------------------------------------------------------
// Rank-select threshold (17th largest), validity, centroids, counts.
// ---------------------------------------------------------------------------
__global__ __launch_bounds__(256) void select_kernel(
    const float* __restrict__ gridv, const float* __restrict__ sA,
    const float* __restrict__ sxA, const float* __restrict__ syA,
    float* __restrict__ nodes, float* __restrict__ validOut, float* __restrict__ cntOut)
{
    __shared__ float sg[256];
    __shared__ float traw;
    __shared__ int cnt;
    const int b = blockIdx.x, i = threadIdx.x;
    const float gv = gridv[b * 256 + i];
    sg[i] = gv;
    if (i == 0) { traw = -1e30f; cnt = 0; }
    __syncthreads();

    int g = 0, e = 0;
    for (int j = 0; j < 256; j++) {
        float o = sg[j];
        g += (o > gv);
        e += (o == gv);
    }
    // sorted-descending index 16 lands inside this value's tie range
    if (g <= 16 && g + e > 16) traw = gv;
    __syncthreads();

    const float thr = fmaxf(traw, 0.9f);
    const bool valid = gv > thr;
    const float s = sA[b * 256 + i];
    const float nx = valid ? sxA[b * 256 + i] / s : 0.0f;
    const float ny = valid ? syA[b * 256 + i] / s : 0.0f;
    nodes[(b * 256 + i) * 2 + 0] = nx;
    nodes[(b * 256 + i) * 2 + 1] = ny;
    validOut[b * 256 + i] = valid ? 1.0f : 0.0f;
    if (valid) atomicAdd(&cnt, 1);
    __syncthreads();
    if (i == 0) cntOut[b] = (float)cnt;
}

// ---------------------------------------------------------------------------
// Launch
// ---------------------------------------------------------------------------
static constexpr size_t align256(size_t v) { return (v + 255) & ~size_t(255); }

extern "C" void kernel_launch(void* const* d_in, const int* in_sizes, int n_in,
                              void* d_out, int out_size, void* d_ws, size_t ws_size,
                              hipStream_t stream)
{
    (void)in_sizes; (void)n_in; (void)out_size; (void)ws_size;
    const float* x   = (const float*)d_in[0];
    const float* w1  = (const float*)d_in[1];
    const float* b1  = (const float*)d_in[2];
    const float* g1  = (const float*)d_in[3];
    const float* be1 = (const float*)d_in[4];
    const float* m1  = (const float*)d_in[5];
    const float* v1  = (const float*)d_in[6];
    const float* w2  = (const float*)d_in[7];
    const float* b2  = (const float*)d_in[8];
    const float* g2  = (const float*)d_in[9];
    const float* be2 = (const float*)d_in[10];
    const float* m2  = (const float*)d_in[11];
    const float* v2  = (const float*)d_in[12];
    const float* w3  = (const float*)d_in[13];
    const float* b3  = (const float*)d_in[14];
    const float* g3  = (const float*)d_in[15];
    const float* be3 = (const float*)d_in[16];
    const float* m3  = (const float*)d_in[17];
    const float* v3  = (const float*)d_in[18];
    const float* w4  = (const float*)d_in[19];
    const float* b4  = (const float*)d_in[20];

    // workspace layout
    constexpr size_t H1_B  = (size_t)NBATCH * OH1 * OW1 * 64 * 2;  // bf16 NHWC
    constexpr size_t H2_B  = (size_t)NBATCH * OH2 * OW2 * 32 * 2;
    constexpr size_t H3_B  = (size_t)NBATCH * OH3 * OW3 * 32 * 2;
    constexpr size_t W2M_B = 32 * 576 * 2;
    constexpr size_t W3M_B = 32 * 288 * 2;
    constexpr size_t SCL_B = 256 * 4;
    constexpr size_t POOL_B = (size_t)NBATCH * 256 * 4;

    char* ws = (char*)d_ws;
    size_t off = 0;
    __bf16* h1  = (__bf16*)(ws + off); off = align256(off + H1_B);
    __bf16* h2  = (__bf16*)(ws + off); off = align256(off + H2_B);
    __bf16* h3  = (__bf16*)(ws + off); off = align256(off + H3_B);
    __bf16* w2m = (__bf16*)(ws + off); off = align256(off + W2M_B);
    __bf16* w3m = (__bf16*)(ws + off); off = align256(off + W3M_B);
    float*  scl = (float*)(ws + off);  off = align256(off + SCL_B);
    float* gridv = (float*)(ws + off); off = align256(off + POOL_B);
    float* sA    = (float*)(ws + off); off = align256(off + POOL_B);
    float* sxA   = (float*)(ws + off); off = align256(off + POOL_B);
    float* syA   = (float*)(ws + off); off = align256(off + POOL_B);

    // d_out layout: nodes[64*256*2] | heat_raw[64*128*128] | valid[64*256] | cnt[64]
    float* out_nodes = (float*)d_out;
    float* out_heat  = out_nodes + (size_t)NBATCH * 256 * 2;
    float* out_valid = out_heat + (size_t)NBATCH * H_IMG * W_IMG;
    float* out_cnt   = out_valid + (size_t)NBATCH * 256;

    prep_kernel<<<64, 256, 0, stream>>>(w2, w3, b1, g1, be1, m1, v1,
                                        b2, g2, be2, m2, v2, b3, g3, be3, m3, v3,
                                        scl, w2m, w3m);

    {
        int total = NBATCH * OH1 * OW1;
        conv1_kernel<<<(total + 255) / 256, 256, 0, stream>>>(x, w1, scl, h1);
    }
    {
        constexpr int NPIX = OW2 * OH2;                   // 17424
        constexpr int TILES = (NPIX + 15) / 16;           // 1089
        dim3 grid((TILES + 7) / 8, NBATCH);
        conv_wmma_kernel<64, OW2, OH2><<<grid, 256, 0, stream>>>(
            h1, w2m, scl + 128, scl + 160, h2);
    }
    {
        constexpr int NPIX = OW3 * OH3;                   // 16900
        constexpr int TILES = (NPIX + 15) / 16;           // 1056 full + tail
        dim3 grid((TILES + 7) / 8, NBATCH);
        conv_wmma_kernel<32, OW3, OH3><<<grid, 256, 0, stream>>>(
            h2, w3m, scl + 192, scl + 224, h3);
    }
    {
        int total = NBATCH * H_IMG * W_IMG;
        conv4_kernel<<<(total + 255) / 256, 256, 0, stream>>>(h3, w4, b4, out_heat);
    }
    pool_kernel<<<NBATCH, 256, 0, stream>>>(out_heat, gridv, sA, sxA, syA);
    select_kernel<<<NBATCH, 256, 0, stream>>>(gridv, sA, sxA, syA,
                                              out_nodes, out_valid, out_cnt);
}
